// EncoderLayer_88510686036847
// MI455X (gfx1250) — compile-verified, matrix operand order checked by reading, not run on hardware
//
#include <hip/hip_runtime.h>
#include <hip/hip_bf16.h>

// ---------------------------------------------------------------------------
// Transformer encoder layer for MI455X (gfx1250), wave32 + WMMA.
// Dense matmuls: v_wmma_f32_16x16x32_f16 with double-buffered LDS tiles.
// Softmax / LayerNorm / residuals in fp32.
// ---------------------------------------------------------------------------

typedef __attribute__((ext_vector_type(16))) _Float16 v16h;
typedef __attribute__((ext_vector_type(8)))  _Float16 v8h;
typedef __attribute__((ext_vector_type(8)))  float    v8f;
typedef __attribute__((ext_vector_type(4)))  float    v4f;

#define D_MODEL 512
#define N_HEADS 8
#define DEPTH   64
#define D_FF    2048
#define SEQ     2048
#define BATCH   2
#define MTOT    (BATCH * SEQ)   // 4096 rows

#define LDS_STRIDE 40           // 32 halves + 8 pad (80B rows: conflict-free frag reads)

// ---------------------------------------------------------------------------
// WMMA 16-bit fragment layout (ISA 7.12.2): lane L: row = L%16, g = L/16,
// element e (0..15): K = k0 + 16*(e/8) + 8*g + (e%8)
// => two contiguous 16-byte chunks at k0+8g and k0+16+8g.
// B uses the mirrored layout, so B is stored transposed (Bt[n][k]).
// ---------------------------------------------------------------------------
__device__ inline v16h frag_from_lds(const _Float16 (*T)[LDS_STRIDE],
                                     int row0, int r, int g) {
  const _Float16* p = &T[row0 + r][8 * g];
  v8h lo = *(const v8h*)p;
  v8h hi = *(const v8h*)(p + 16);
  v16h f;
#pragma unroll
  for (int i = 0; i < 8; ++i) { f[i] = lo[i]; f[i + 8] = hi[i]; }
  return f;
}

// ---------------------------------------------------------------------------
// Generic batched WMMA GEMM:  C = A(MxK) * Bt(NxK)^T  (+ bias)
// Block: 256 threads = 8 waves (2 in M x 4 in N); block tile 128x128;
// wave tile 64x32 = 4x2 accumulators. A/B tiles (128x32 f16 each) staged in
// LDS, double buffered: global loads for step s+1 overlap WMMAs of step s.
// Epilogues:
//   0 EPI_F32      : float C[m*ldc+n] = acc + bias
//   1 EPI_F16RELU  : _Float16 C = max(acc+bias, 0)
//   2 EPI_SPLIT    : head-split f16  [b,h,s,d] = (acc+bias)*scale   (Q/K)
//   3 EPI_SPLIT_T  : head-split f16  [b,h,d,s] = acc+bias           (V^T)
// ---------------------------------------------------------------------------
#define EPI_F32     0
#define EPI_F16RELU 1
#define EPI_SPLIT   2
#define EPI_SPLIT_T 3

template <int EPI>
__global__ __launch_bounds__(256) void gemm_wmma_kernel(
    const _Float16* __restrict__ A,  long long sA, int lda,
    const _Float16* __restrict__ Bt, long long sB, int ldb,
    const float* __restrict__ bias,
    void* __restrict__ Cout, long long sC, int ldc,
    int K, float scale) {
  __shared__ _Float16 Ash[2][128][LDS_STRIDE];
  __shared__ _Float16 Bsh[2][128][LDS_STRIDE];

  const int z = blockIdx.z;
  A  += (size_t)z * (size_t)sA;
  Bt += (size_t)z * (size_t)sB;

  const int tid  = threadIdx.x;
  const int wave = tid >> 5;
  const int lane = tid & 31;
  const int r = lane & 15, g = lane >> 4;
  const int wm = wave >> 2;            // 0..1
  const int wn = wave & 3;             // 0..3
  const int rowBlock0 = blockIdx.y * 128;      // block base row
  const int colBlock0 = blockIdx.x * 128;      // block base col
  const int rowW = wm * 64;                    // wave row within block
  const int colW = wn * 32;                    // wave col within block

  // Per-thread staging slots: idx = tid + 256*it in [0,512): row=idx>>2, c=idx&3
  const int ldRow0 = tid >> 2, ldC0 = tid & 3;
  const int ldRow1 = (tid + 256) >> 2, ldC1 = (tid + 256) & 3;

  v8f acc[4][2] = {};
  v8h aR[2], bR[2];

  // Prologue: load k-step 0 tiles into buffer 0.
  aR[0] = *(const v8h*)(A  + (size_t)(rowBlock0 + ldRow0) * lda + 8 * ldC0);
  aR[1] = *(const v8h*)(A  + (size_t)(rowBlock0 + ldRow1) * lda + 8 * ldC1);
  bR[0] = *(const v8h*)(Bt + (size_t)(colBlock0 + ldRow0) * ldb + 8 * ldC0);
  bR[1] = *(const v8h*)(Bt + (size_t)(colBlock0 + ldRow1) * ldb + 8 * ldC1);
  *(v8h*)(&Ash[0][ldRow0][8 * ldC0]) = aR[0];
  *(v8h*)(&Ash[0][ldRow1][8 * ldC1]) = aR[1];
  *(v8h*)(&Bsh[0][ldRow0][8 * ldC0]) = bR[0];
  *(v8h*)(&Bsh[0][ldRow1][8 * ldC1]) = bR[1];
  __syncthreads();

  const int nsteps = K >> 5;
  for (int s = 0; s < nsteps; ++s) {
    const int buf = s & 1;
    const bool hasNext = (s + 1) < nsteps;
    if (hasNext) {
      const int k0 = (s + 1) << 5;
      aR[0] = *(const v8h*)(A  + (size_t)(rowBlock0 + ldRow0) * lda + k0 + 8 * ldC0);
      aR[1] = *(const v8h*)(A  + (size_t)(rowBlock0 + ldRow1) * lda + k0 + 8 * ldC1);
      bR[0] = *(const v8h*)(Bt + (size_t)(colBlock0 + ldRow0) * ldb + k0 + 8 * ldC0);
      bR[1] = *(const v8h*)(Bt + (size_t)(colBlock0 + ldRow1) * ldb + k0 + 8 * ldC1);
    }

    v16h a[4], b[2];
#pragma unroll
    for (int i = 0; i < 4; ++i) a[i] = frag_from_lds(Ash[buf], rowW + 16 * i, r, g);
#pragma unroll
    for (int j = 0; j < 2; ++j) b[j] = frag_from_lds(Bsh[buf], colW + 16 * j, r, g);
#pragma unroll
    for (int i = 0; i < 4; ++i)
#pragma unroll
      for (int j = 0; j < 2; ++j)
        acc[i][j] = __builtin_amdgcn_wmma_f32_16x16x32_f16(
            false, a[i], false, b[j], (short)0, acc[i][j], false, false);

    if (hasNext) {
      const int nb = buf ^ 1;
      *(v8h*)(&Ash[nb][ldRow0][8 * ldC0]) = aR[0];
      *(v8h*)(&Ash[nb][ldRow1][8 * ldC1]) = aR[1];
      *(v8h*)(&Bsh[nb][ldRow0][8 * ldC0]) = bR[0];
      *(v8h*)(&Bsh[nb][ldRow1][8 * ldC1]) = bR[1];
      __syncthreads();
    }
  }

  // Epilogue. C/D layout: element v of v8f -> m = 16*i + 8*g + v, n = 16*j + r.
#pragma unroll
  for (int i = 0; i < 4; ++i) {
#pragma unroll
    for (int j = 0; j < 2; ++j) {
      const int n = colBlock0 + colW + 16 * j + r;
      const float bz = bias ? bias[n] : 0.0f;
#pragma unroll
      for (int v = 0; v < 8; ++v) {
        const int m = rowBlock0 + rowW + 16 * i + 8 * g + v;
        float val = acc[i][j][v] + bz;
        if (EPI == EPI_F32) {
          ((float*)Cout)[(size_t)z * (size_t)sC + (size_t)m * ldc + n] = val;
        } else if (EPI == EPI_F16RELU) {
          val = fmaxf(val, 0.0f);
          ((_Float16*)Cout)[(size_t)m * ldc + n] = (_Float16)val;
        } else if (EPI == EPI_SPLIT) {
          const int bb = m >> 11, ss = m & 2047;
          const int h = n >> 6, d = n & 63;
          ((_Float16*)Cout)[(((size_t)(bb * N_HEADS + h) * SEQ + ss) * DEPTH) + d] =
              (_Float16)(val * scale);
        } else { // EPI_SPLIT_T
          const int bb = m >> 11, ss = m & 2047;
          const int h = n >> 6, d = n & 63;
          ((_Float16*)Cout)[(((size_t)(bb * N_HEADS + h) * DEPTH + d) * SEQ) + ss] =
              (_Float16)val;
        }
      }
    }
  }
}

// ---------------------------------------------------------------------------
// Context GEMM: per (b,h): Ctx = P(2048x2048 fp32) @ V(2048x64).
// P tile converted fp32->f16 in registers, staged in LDS (each HBM byte of the
// 268MB attention matrix is read exactly once). Vt (f16, [d][t]) tile in LDS.
// Block: 256 threads = 8 waves (4 in M x 2 in N); tile 128x64; wave 32x32.
// ---------------------------------------------------------------------------
__global__ __launch_bounds__(256) void gemm_ctx_kernel(
    const float* __restrict__ attn, const _Float16* __restrict__ Vt,
    _Float16* __restrict__ Ctxh) {
  __shared__ _Float16 Ash[2][128][LDS_STRIDE];
  __shared__ _Float16 Bsh[2][64][LDS_STRIDE];

  const int z = blockIdx.z;                       // b*8 + h
  const float*    Ag = attn + (size_t)z * SEQ * SEQ;
  const _Float16* Bg = Vt   + (size_t)z * DEPTH * SEQ;

  const int tid  = threadIdx.x;
  const int wave = tid >> 5;
  const int lane = tid & 31;
  const int r = lane & 15, g = lane >> 4;
  const int wm = wave >> 1;            // 0..3
  const int wn = wave & 1;             // 0..1
  const int rowBlock0 = blockIdx.y * 128;
  const int rowW = wm * 32;
  const int colW = wn * 32;

  // A staging: 512 chunks of 8 halves -> 2 per thread (convert f32->f16).
  const int aRow0 = tid >> 2, aC0 = tid & 3;
  const int aRow1 = (tid + 256) >> 2, aC1 = (tid + 256) & 3;
  // B staging: 256 chunks (64 rows x 4) -> 1 per thread.
  const int bRow = tid >> 2, bC = tid & 3;

  v8h aR[2], bR;

  auto loadA = [&](int k0, int row, int c) -> v8h {
    const float* p = Ag + (size_t)(rowBlock0 + row) * SEQ + k0 + 8 * c;
    v4f f0 = *(const v4f*)(p + 0);
    v4f f1 = *(const v4f*)(p + 4);
    v8h h;
#pragma unroll
    for (int i = 0; i < 4; ++i) { h[i] = (_Float16)f0[i]; h[i + 4] = (_Float16)f1[i]; }
    return h;
  };

  // Prologue: k-step 0 into buffer 0.
  aR[0] = loadA(0, aRow0, aC0);
  aR[1] = loadA(0, aRow1, aC1);
  bR    = *(const v8h*)(Bg + (size_t)bRow * SEQ + 8 * bC);
  *(v8h*)(&Ash[0][aRow0][8 * aC0]) = aR[0];
  *(v8h*)(&Ash[0][aRow1][8 * aC1]) = aR[1];
  *(v8h*)(&Bsh[0][bRow][8 * bC])   = bR;
  __syncthreads();

  v8f acc[2][2] = {};
  const int nsteps = SEQ >> 5;
  for (int s = 0; s < nsteps; ++s) {
    const int buf = s & 1;
    const bool hasNext = (s + 1) < nsteps;
    if (hasNext) {
      const int k0 = (s + 1) << 5;
      aR[0] = loadA(k0, aRow0, aC0);
      aR[1] = loadA(k0, aRow1, aC1);
      bR    = *(const v8h*)(Bg + (size_t)bRow * SEQ + k0 + 8 * bC);
    }

    v16h a[2], b[2];
#pragma unroll
    for (int i = 0; i < 2; ++i) a[i] = frag_from_lds(Ash[buf], rowW + 16 * i, r, g);
#pragma unroll
    for (int j = 0; j < 2; ++j) b[j] = frag_from_lds(Bsh[buf], colW + 16 * j, r, g);
#pragma unroll
    for (int i = 0; i < 2; ++i)
#pragma unroll
      for (int j = 0; j < 2; ++j)
        acc[i][j] = __builtin_amdgcn_wmma_f32_16x16x32_f16(
            false, a[i], false, b[j], (short)0, acc[i][j], false, false);

    if (hasNext) {
      const int nb = buf ^ 1;
      *(v8h*)(&Ash[nb][aRow0][8 * aC0]) = aR[0];
      *(v8h*)(&Ash[nb][aRow1][8 * aC1]) = aR[1];
      *(v8h*)(&Bsh[nb][bRow][8 * bC])   = bR;
      __syncthreads();
    }
  }

  const int bb = z >> 3, h = z & 7;
#pragma unroll
  for (int i = 0; i < 2; ++i)
#pragma unroll
    for (int j = 0; j < 2; ++j) {
      const int n = colW + 16 * j + r;   // d within head
#pragma unroll
      for (int v = 0; v < 8; ++v) {
        const int m = rowBlock0 + rowW + 16 * i + 8 * g + v;  // s
        Ctxh[((size_t)(bb * SEQ + m)) * D_MODEL + h * DEPTH + n] =
            (_Float16)acc[i][j][v];
      }
    }
}

// ---------------------------------------------------------------------------
// Reductions
// ---------------------------------------------------------------------------
__device__ inline float waveSum(float v) {
#pragma unroll
  for (int o = 16; o > 0; o >>= 1) v += __shfl_xor(v, o, 32);
  return v;
}
__device__ inline float waveMax(float v) {
#pragma unroll
  for (int o = 16; o > 0; o >>= 1) v = fmaxf(v, __shfl_xor(v, o, 32));
  return v;
}

// Row softmax over 2048 fp32 values, in place. One block (256 thr) per row.
__global__ __launch_bounds__(256) void softmax_kernel(float* __restrict__ attn) {
  float* p = attn + (size_t)blockIdx.x * SEQ;
  const int tid = threadIdx.x;
  float v[8];
  float mx = -3.0e38f;
#pragma unroll
  for (int i = 0; i < 8; ++i) { v[i] = p[tid + 256 * i]; mx = fmaxf(mx, v[i]); }

  __shared__ float red[8];
  const int wave = tid >> 5, lane = tid & 31;
  mx = waveMax(mx);
  if (lane == 0) red[wave] = mx;
  __syncthreads();
  float bmx = red[0];
#pragma unroll
  for (int w = 1; w < 8; ++w) bmx = fmaxf(bmx, red[w]);
  __syncthreads();

  float sum = 0.0f;
#pragma unroll
  for (int i = 0; i < 8; ++i) { v[i] = __expf(v[i] - bmx); sum += v[i]; }
  sum = waveSum(sum);
  if (lane == 0) red[wave] = sum;
  __syncthreads();
  float bsum = 0.0f;
#pragma unroll
  for (int w = 0; w < 8; ++w) bsum += red[w];
  const float inv = 1.0f / bsum;
#pragma unroll
  for (int i = 0; i < 8; ++i) p[tid + 256 * i] = v[i] * inv;
}

// Residual + LayerNorm over D_MODEL=512: out = LN(A+B)*g+b.
// Writes fp32 (outF) and optionally f16 (outH). One block (256 thr) per row.
__global__ __launch_bounds__(256) void ln_residual_kernel(
    const float* __restrict__ A, const float* __restrict__ B,
    const float* __restrict__ gamma, const float* __restrict__ beta,
    float* __restrict__ outF, _Float16* __restrict__ outH, int writeHalf) {
  const size_t row = blockIdx.x;
  const float* pa = A + row * D_MODEL;
  const float* pb = B + row * D_MODEL;
  const int tid = threadIdx.x;

  const float v0 = pa[tid] + pb[tid];
  const float v1 = pa[tid + 256] + pb[tid + 256];
  float s = v0 + v1;
  float q = v0 * v0 + v1 * v1;

  __shared__ float ss[8], qq[8];
  const int wave = tid >> 5, lane = tid & 31;
  s = waveSum(s); q = waveSum(q);
  if (lane == 0) { ss[wave] = s; qq[wave] = q; }
  __syncthreads();
  float ts = 0.0f, tq = 0.0f;
#pragma unroll
  for (int w = 0; w < 8; ++w) { ts += ss[w]; tq += qq[w]; }
  const float mu  = ts * (1.0f / D_MODEL);
  const float var = tq * (1.0f / D_MODEL) - mu * mu;
  const float rstd = rsqrtf(var + 1e-5f);

  const float y0 = (v0 - mu) * rstd * gamma[tid] + beta[tid];
  const float y1 = (v1 - mu) * rstd * gamma[tid + 256] + beta[tid + 256];
  outF[row * D_MODEL + tid]       = y0;
  outF[row * D_MODEL + tid + 256] = y1;
  if (writeHalf) {
    outH[row * D_MODEL + tid]       = (_Float16)y0;
    outH[row * D_MODEL + tid + 256] = (_Float16)y1;
  }
}

// fp32 -> f16 elementwise
__global__ void cvt_f16_kernel(const float* __restrict__ in,
                               _Float16* __restrict__ out, int n) {
  const int i = blockIdx.x * 256 + threadIdx.x;
  if (i < n) out[i] = (_Float16)in[i];
}

// Wt[n*K + k] = (f16) W[k*N + n]   (write-coalesced transpose+convert)
__global__ void transpose_cvt_kernel(const float* __restrict__ W,
                                     _Float16* __restrict__ Wt, int K, int N) {
  const size_t idx = (size_t)blockIdx.x * 256 + threadIdx.x;
  if (idx >= (size_t)K * N) return;
  const int n = (int)(idx / K);
  const int k = (int)(idx % K);
  Wt[idx] = (_Float16)W[(size_t)k * N + n];
}

// ---------------------------------------------------------------------------
// Host-side orchestration
// ---------------------------------------------------------------------------
extern "C" void kernel_launch(void* const* d_in, const int* in_sizes, int n_in,
                              void* d_out, int out_size, void* d_ws, size_t ws_size,
                              hipStream_t stream) {
  const float* x    = (const float*)d_in[0];
  const float* wq   = (const float*)d_in[1];
  const float* bq   = (const float*)d_in[2];
  const float* wk   = (const float*)d_in[3];
  const float* bk   = (const float*)d_in[4];
  const float* wv   = (const float*)d_in[5];
  const float* bv   = (const float*)d_in[6];
  const float* wo   = (const float*)d_in[7];
  const float* bo   = (const float*)d_in[8];
  const float* w1   = (const float*)d_in[9];
  const float* b1   = (const float*)d_in[10];
  const float* w2   = (const float*)d_in[11];
  const float* b2   = (const float*)d_in[12];
  const float* ln1g = (const float*)d_in[13];
  const float* ln1b = (const float*)d_in[14];
  const float* ln2g = (const float*)d_in[15];
  const float* ln2b = (const float*)d_in[16];

  float* outPtr  = (float*)d_out;                                  // [2,2048,512]
  float* attnPtr = outPtr + (size_t)BATCH * SEQ * D_MODEL;         // [2,8,2048,2048]

  // Workspace carve-out (256B aligned blocks)
  char* wsp = (char*)d_ws;
  auto carve = [&](size_t bytes) -> void* {
    void* p = (void*)wsp;
    wsp += (bytes + 255) & ~(size_t)255;
    return p;
  };
  _Float16* Xh   = (_Float16*)carve((size_t)MTOT * D_MODEL * 2);
  _Float16* Wqt  = (_Float16*)carve((size_t)D_MODEL * D_MODEL * 2);
  _Float16* Wkt  = (_Float16*)carve((size_t)D_MODEL * D_MODEL * 2);
  _Float16* Wvt  = (_Float16*)carve((size_t)D_MODEL * D_MODEL * 2);
  _Float16* Wot  = (_Float16*)carve((size_t)D_MODEL * D_MODEL * 2);
  _Float16* W1t  = (_Float16*)carve((size_t)D_FF * D_MODEL * 2);   // [N=2048][K=512]
  _Float16* W2t  = (_Float16*)carve((size_t)D_MODEL * D_FF * 2);   // [N=512][K=2048]
  _Float16* Qh   = (_Float16*)carve((size_t)MTOT * D_MODEL * 2);   // [b,h,s,d]
  _Float16* Kh   = (_Float16*)carve((size_t)MTOT * D_MODEL * 2);   // [b,h,s,d]
  _Float16* Vt   = (_Float16*)carve((size_t)MTOT * D_MODEL * 2);   // [b,h,d,s]
  _Float16* Ctxh = (_Float16*)carve((size_t)MTOT * D_MODEL * 2);
  float*    AttnOutF = (float*)carve((size_t)MTOT * D_MODEL * 4);
  float*    Hf   = (float*)carve((size_t)MTOT * D_MODEL * 4);
  _Float16* Hh   = (_Float16*)carve((size_t)MTOT * D_MODEL * 2);
  _Float16* F1h  = (_Float16*)carve((size_t)MTOT * D_FF * 2);
  float*    FF2f = (float*)carve((size_t)MTOT * D_MODEL * 4);

  const dim3 blk(256);

  // ---- precision conversion ----
  {
    const int n = MTOT * D_MODEL;
    cvt_f16_kernel<<<(n + 255) / 256, blk, 0, stream>>>(x, Xh, n);
  }
  transpose_cvt_kernel<<<(D_MODEL * D_MODEL + 255) / 256, blk, 0, stream>>>(wq, Wqt, D_MODEL, D_MODEL);
  transpose_cvt_kernel<<<(D_MODEL * D_MODEL + 255) / 256, blk, 0, stream>>>(wk, Wkt, D_MODEL, D_MODEL);
  transpose_cvt_kernel<<<(D_MODEL * D_MODEL + 255) / 256, blk, 0, stream>>>(wv, Wvt, D_MODEL, D_MODEL);
  transpose_cvt_kernel<<<(D_MODEL * D_MODEL + 255) / 256, blk, 0, stream>>>(wo, Wot, D_MODEL, D_MODEL);
  transpose_cvt_kernel<<<(D_MODEL * D_FF + 255) / 256, blk, 0, stream>>>(w1, W1t, D_MODEL, D_FF); // K=512,N=2048
  transpose_cvt_kernel<<<(D_FF * D_MODEL + 255) / 256, blk, 0, stream>>>(w2, W2t, D_FF, D_MODEL); // K=2048,N=512

  // ---- QKV projections (head-split epilogues; Q folded with 1/sqrt(depth)) ----
  {
    dim3 grid(D_MODEL / 128, MTOT / 128, 1);
    gemm_wmma_kernel<EPI_SPLIT><<<grid, blk, 0, stream>>>(
        Xh, 0, D_MODEL, Wqt, 0, D_MODEL, bq, Qh, 0, 0, D_MODEL, 0.125f);
    gemm_wmma_kernel<EPI_SPLIT><<<grid, blk, 0, stream>>>(
        Xh, 0, D_MODEL, Wkt, 0, D_MODEL, bk, Kh, 0, 0, D_MODEL, 1.0f);
    gemm_wmma_kernel<EPI_SPLIT_T><<<grid, blk, 0, stream>>>(
        Xh, 0, D_MODEL, Wvt, 0, D_MODEL, bv, Vt, 0, 0, D_MODEL, 1.0f);
  }

  // ---- attention scores: per (b,h) Q @ K^T, fp32 straight into d_out ----
  {
    dim3 grid(SEQ / 128, SEQ / 128, BATCH * N_HEADS);
    gemm_wmma_kernel<EPI_F32><<<grid, blk, 0, stream>>>(
        Qh, (long long)SEQ * DEPTH, DEPTH,
        Kh, (long long)SEQ * DEPTH, DEPTH,
        nullptr,
        attnPtr, (long long)SEQ * SEQ, SEQ,
        DEPTH, 1.0f);
  }

  // ---- softmax in place on attention ----
  softmax_kernel<<<BATCH * N_HEADS * SEQ, blk, 0, stream>>>(attnPtr);

  // ---- context = P @ V (fp32 P converted on the fly, f16 merged-head out) ----
  {
    dim3 grid(1, SEQ / 128, BATCH * N_HEADS);
    gemm_ctx_kernel<<<grid, blk, 0, stream>>>(attnPtr, Vt, Ctxh);
  }

  // ---- output projection ----
  {
    dim3 grid(D_MODEL / 128, MTOT / 128, 1);
    gemm_wmma_kernel<EPI_F32><<<grid, blk, 0, stream>>>(
        Ctxh, 0, D_MODEL, Wot, 0, D_MODEL, bo,
        AttnOutF, 0, D_MODEL, D_MODEL, 1.0f);
  }

  // ---- h = LN1(x + attn_out); fp32 + f16 views ----
  ln_residual_kernel<<<MTOT, blk, 0, stream>>>(x, AttnOutF, ln1g, ln1b, Hf, Hh, 1);

  // ---- FFN ----
  {
    dim3 grid(D_FF / 128, MTOT / 128, 1);
    gemm_wmma_kernel<EPI_F16RELU><<<grid, blk, 0, stream>>>(
        Hh, 0, D_MODEL, W1t, 0, D_MODEL, b1,
        F1h, 0, D_FF, D_MODEL, 1.0f);
  }
  {
    dim3 grid(D_MODEL / 128, MTOT / 128, 1);
    gemm_wmma_kernel<EPI_F32><<<grid, blk, 0, stream>>>(
        F1h, 0, D_FF, W2t, 0, D_FF, b2,
        FF2f, 0, D_MODEL, D_FF, 1.0f);
  }

  // ---- out = LN2(h + ff) ----
  ln_residual_kernel<<<MTOT, blk, 0, stream>>>(Hf, FF2f, ln2g, ln2b, outPtr, nullptr, 0);

  (void)in_sizes; (void)n_in; (void)out_size; (void)ws_size;
}
